// StandardTopKMoE_49821620634172
// MI455X (gfx1250) — matrix operand order repaired and checked by reading, not compiled, not run on hardware
//
#include <hip/hip_runtime.h>

typedef __attribute__((ext_vector_type(16))) __bf16 v16bf;
typedef __attribute__((ext_vector_type(8)))  __bf16 v8bf;
typedef __attribute__((ext_vector_type(8)))  float  v8f;

union BFrag { v16bf v; v8bf h[2]; };

#define LDSS 40   // ushort stride per LDS row (80 bytes): 16B-aligned chunks, conflict padding

__device__ __forceinline__ unsigned short f2bf(float f) {
  unsigned int u = __float_as_uint(f);
  u += 0x7FFFu + ((u >> 16) & 1u);          // round-to-nearest-even
  return (unsigned short)(u >> 16);
}

// ---------------------------------------------------------------- zero output
__global__ void zero_f32_kernel(float4* __restrict__ p, size_t n4) {
  size_t i = (size_t)blockIdx.x * blockDim.x + threadIdx.x;
  size_t stride = (size_t)gridDim.x * blockDim.x;
  float4 z; z.x = z.y = z.z = z.w = 0.f;
  for (; i < n4; i += stride) p[i] = z;
}

// ---------------------------------------------------------------- f32 -> bf16 (flat)
__global__ void cvt_bf16_kernel(const float* __restrict__ src,
                                unsigned short* __restrict__ dst, size_t n4) {
  size_t i = (size_t)blockIdx.x * blockDim.x + threadIdx.x;
  size_t stride = (size_t)gridDim.x * blockDim.x;
  const float4* s4 = (const float4*)src;
  ushort4* d4 = (ushort4*)dst;
  for (; i < n4; i += stride) {
    float4 v = s4[i];
    ushort4 o;
    o.x = f2bf(v.x); o.y = f2bf(v.y); o.z = f2bf(v.z); o.w = f2bf(v.w);
    d4[i] = o;
  }
}

// ---------------------------------------------------------------- f32 [R,C] -> bf16 [C,R]
__global__ __launch_bounds__(256)
void transpose_cvt_kernel(const float* __restrict__ src,
                          unsigned short* __restrict__ dst, int R, int C) {
  __shared__ float tile[32][33];
  const int tx = threadIdx.x, ty = threadIdx.y;
  const int r0 = blockIdx.y * 32, c0 = blockIdx.x * 32;
#pragma unroll
  for (int j = 0; j < 32; j += 8)
    tile[ty + j][tx] = src[(size_t)(r0 + ty + j) * C + (c0 + tx)];
  __syncthreads();
#pragma unroll
  for (int j = 0; j < 32; j += 8)
    dst[(size_t)(c0 + ty + j) * R + (r0 + tx)] = f2bf(tile[tx][ty + j]);
}

// ---------------------------------------------------------------- router: softmax top-2 weights
// one wave32 per token; writes dense-but-sparse weight table Wexp[token][8]
__global__ __launch_bounds__(256)
void router_kernel(const float* __restrict__ X, const float* __restrict__ Wr,
                   float* __restrict__ Wexp, int D) {
  const int lane = threadIdx.x & 31;
  const int token = blockIdx.x * 8 + (threadIdx.x >> 5);
  const float* xr = X + (size_t)token * D;
  float a[8];
#pragma unroll
  for (int e = 0; e < 8; ++e) a[e] = 0.f;
  for (int d = lane; d < D; d += 32) {
    float xv = xr[d];
    const float* wr = Wr + (size_t)d * 8;
#pragma unroll
    for (int e = 0; e < 8; ++e) a[e] += xv * wr[e];
  }
#pragma unroll
  for (int off = 16; off >= 1; off >>= 1) {
#pragma unroll
    for (int e = 0; e < 8; ++e) a[e] += __shfl_xor(a[e], off, 32);
  }
  if (lane == 0) {
    int i1 = 0;
#pragma unroll
    for (int e = 1; e < 8; ++e) if (a[e] > a[i1]) i1 = e;
    int i2 = (i1 == 0) ? 1 : 0;
#pragma unroll
    for (int e = 0; e < 8; ++e) if (e != i1 && a[e] > a[i2]) i2 = e;
    // softmax -> top-2 -> renormalize == 2-way softmax over the two winners
    float w1 = 1.f / (1.f + expf(a[i2] - a[i1]));
    float* o = Wexp + (size_t)token * 8;
#pragma unroll
    for (int e = 0; e < 8; ++e) o[e] = 0.f;
    o[i1] = w1;
    o[i2] = 1.f - w1;
  }
}

// ---------------------------------------------------------------- fused gate+up GEMM
// H[M,N] = silu(Xb @ WgT^T) * (Xb @ WuT^T);  Xb:[M,K] bf16, WgT/WuT:[N,K] bf16
// 128x128 block tile, 8 waves each 64x32, LDS double-buffered K stages of 32.
__global__ __launch_bounds__(256)
void moe_gateup_kernel(const unsigned short* __restrict__ Xb,
                       const unsigned short* __restrict__ WgT,
                       const unsigned short* __restrict__ WuT,
                       unsigned short* __restrict__ H,
                       int M, int N, int K) {
  __shared__ unsigned short sA[2][128 * LDSS];
  __shared__ unsigned short sG[2][128 * LDSS];
  __shared__ unsigned short sU[2][128 * LDSS];

  const int tid = threadIdx.x;
  const int lane = tid & 31;
  const int wave = tid >> 5;
  const int wm = wave >> 2;      // 0..1 -> 64-row half
  const int wn = wave & 3;       // 0..3 -> 32-col strip
  const int l16 = lane & 15;
  const int lh  = lane >> 4;     // half-wave select

  const int m0 = blockIdx.y * 128;
  const int n0 = blockIdx.x * 128;

  const v8f vzero = {0.f,0.f,0.f,0.f,0.f,0.f,0.f,0.f};
  v8f accG[4][2], accU[4][2];
#pragma unroll
  for (int i = 0; i < 4; ++i)
#pragma unroll
    for (int j = 0; j < 2; ++j) { accG[i][j] = vzero; accU[i][j] = vzero; }

  // global->LDS tile loaders: each thread moves two 16B chunks per tile
  const int lr = tid >> 2;          // 0..63
  const int lk = (tid & 3) * 8;     // elem col: 0,8,16,24
  const int ldsOff0 = lr * LDSS + lk;
  const int ldsOff1 = (lr + 64) * LDSS + lk;
  const size_t aRow0 = (size_t)(m0 + lr) * K + lk;
  const size_t aRow1 = aRow0 + (size_t)64 * K;
  const size_t bRow0 = (size_t)(n0 + lr) * K + lk;
  const size_t bRow1 = bRow0 + (size_t)64 * K;

  // prologue: stage 0 -> buffer 0
  {
    uint4 ra0 = *(const uint4*)(Xb  + aRow0);
    uint4 ra1 = *(const uint4*)(Xb  + aRow1);
    uint4 rg0 = *(const uint4*)(WgT + bRow0);
    uint4 rg1 = *(const uint4*)(WgT + bRow1);
    uint4 ru0 = *(const uint4*)(WuT + bRow0);
    uint4 ru1 = *(const uint4*)(WuT + bRow1);
    *(uint4*)(&sA[0][ldsOff0]) = ra0;
    *(uint4*)(&sA[0][ldsOff1]) = ra1;
    *(uint4*)(&sG[0][ldsOff0]) = rg0;
    *(uint4*)(&sG[0][ldsOff1]) = rg1;
    *(uint4*)(&sU[0][ldsOff0]) = ru0;
    *(uint4*)(&sU[0][ldsOff1]) = ru1;
  }

  const int S = K >> 5;   // stages of 32
  for (int s = 0; s < S; ++s) {
    const int cur = s & 1;
    __syncthreads();                          // buf[cur] ready; buf[cur^1] free

    // stage s+1: global -> registers (in flight under the WMMAs below)
    uint4 ra0, ra1, rg0, rg1, ru0, ru1;
    const bool more = (s + 1) < S;
    if (more) {
      const int kt = (s + 1) << 5;
      ra0 = *(const uint4*)(Xb  + aRow0 + kt);
      ra1 = *(const uint4*)(Xb  + aRow1 + kt);
      rg0 = *(const uint4*)(WgT + bRow0 + kt);
      rg1 = *(const uint4*)(WgT + bRow1 + kt);
      ru0 = *(const uint4*)(WuT + bRow0 + kt);
      ru1 = *(const uint4*)(WuT + bRow1 + kt);
    }

    // fragments from buf[cur] + 16 WMMAs
    BFrag af[4], bg[2], bu[2];
#pragma unroll
    for (int nt = 0; nt < 2; ++nt) {
      const unsigned short* pg = &sG[cur][(wn * 32 + nt * 16 + l16) * LDSS + 16 * lh];
      bg[nt].h[0] = *(const v8bf*)(pg);
      bg[nt].h[1] = *(const v8bf*)(pg + 8);
      const unsigned short* pu = &sU[cur][(wn * 32 + nt * 16 + l16) * LDSS + 16 * lh];
      bu[nt].h[0] = *(const v8bf*)(pu);
      bu[nt].h[1] = *(const v8bf*)(pu + 8);
    }
#pragma unroll
    for (int mt = 0; mt < 4; ++mt) {
      const unsigned short* base = &sA[cur][(wm * 64 + mt * 16 + l16) * LDSS];
      af[mt].h[0] = *(const v8bf*)(base + 16 * lh);       // K = 16*lh .. +8
      af[mt].h[1] = *(const v8bf*)(base + 16 * lh + 8);   // K = 16*lh+8 .. +8
#pragma unroll
      for (int nt = 0; nt < 2; ++nt) {
        accG[mt][nt] = __builtin_amdgcn_wmma_f32_16x16x32_bf16(
            false, af[mt].v, false, bg[nt].v, (short)0, accG[mt][nt], false, false);
        accU[mt][nt] = __builtin_amdgcn_wmma_f32_16x16x32_bf16(
            false, af[mt].v, false, bu[nt].v, (short)0, accU[mt][nt], false, false);
      }
    }

    // park stage s+1 into buf[cur^1]
    if (more) {
      const int nxt = cur ^ 1;
      *(uint4*)(&sA[nxt][ldsOff0]) = ra0;
      *(uint4*)(&sA[nxt][ldsOff1]) = ra1;
      *(uint4*)(&sG[nxt][ldsOff0]) = rg0;
      *(uint4*)(&sG[nxt][ldsOff1]) = rg1;
      *(uint4*)(&sU[nxt][ldsOff0]) = ru0;
      *(uint4*)(&sU[nxt][ldsOff1]) = ru1;
    }
  }

  // epilogue: h = silu(g) * u, store bf16
#pragma unroll
  for (int mt = 0; mt < 4; ++mt) {
#pragma unroll
    for (int nt = 0; nt < 2; ++nt) {
      const int col = n0 + wn * 32 + nt * 16 + l16;
#pragma unroll
      for (int v = 0; v < 8; ++v) {
        const int row = m0 + wm * 64 + mt * 16 + lh * 8 + v;
        float g = accG[mt][nt][v];
        float u = accU[mt][nt][v];
        float si = g / (1.f + __expf(-g));
        H[(size_t)row * N + col] = f2bf(si * u);
      }
    }
  }
}

// ---------------------------------------------------------------- down-proj GEMM + routed accumulate
// Out[M,N] += wexp[row][e] * (Hb @ WdT^T);  Hb:[M,K] bf16, WdT:[N,K] bf16
__global__ __launch_bounds__(256)
void moe_down_kernel(const unsigned short* __restrict__ Hb,
                     const unsigned short* __restrict__ WdT,
                     const float* __restrict__ Wexp,
                     float* __restrict__ Out,
                     int M, int N, int K, int expert) {
  __shared__ unsigned short sA[2][128 * LDSS];
  __shared__ unsigned short sB[2][128 * LDSS];

  const int tid = threadIdx.x;
  const int lane = tid & 31;
  const int wave = tid >> 5;
  const int wm = wave >> 2;
  const int wn = wave & 3;
  const int l16 = lane & 15;
  const int lh  = lane >> 4;

  const int m0 = blockIdx.y * 128;
  const int n0 = blockIdx.x * 128;

  const v8f vzero = {0.f,0.f,0.f,0.f,0.f,0.f,0.f,0.f};
  v8f acc[4][2];
#pragma unroll
  for (int i = 0; i < 4; ++i)
#pragma unroll
    for (int j = 0; j < 2; ++j) acc[i][j] = vzero;

  const int lr = tid >> 2;
  const int lk = (tid & 3) * 8;
  const int ldsOff0 = lr * LDSS + lk;
  const int ldsOff1 = (lr + 64) * LDSS + lk;
  const size_t aRow0 = (size_t)(m0 + lr) * K + lk;
  const size_t aRow1 = aRow0 + (size_t)64 * K;
  const size_t bRow0 = (size_t)(n0 + lr) * K + lk;
  const size_t bRow1 = bRow0 + (size_t)64 * K;

  {
    uint4 ra0 = *(const uint4*)(Hb  + aRow0);
    uint4 ra1 = *(const uint4*)(Hb  + aRow1);
    uint4 rb0 = *(const uint4*)(WdT + bRow0);
    uint4 rb1 = *(const uint4*)(WdT + bRow1);
    *(uint4*)(&sA[0][ldsOff0]) = ra0;
    *(uint4*)(&sA[0][ldsOff1]) = ra1;
    *(uint4*)(&sB[0][ldsOff0]) = rb0;
    *(uint4*)(&sB[0][ldsOff1]) = rb1;
  }

  const int S = K >> 5;
  for (int s = 0; s < S; ++s) {
    const int cur = s & 1;
    __syncthreads();

    uint4 ra0, ra1, rb0, rb1;
    const bool more = (s + 1) < S;
    if (more) {
      const int kt = (s + 1) << 5;
      ra0 = *(const uint4*)(Hb  + aRow0 + kt);
      ra1 = *(const uint4*)(Hb  + aRow1 + kt);
      rb0 = *(const uint4*)(WdT + bRow0 + kt);
      rb1 = *(const uint4*)(WdT + bRow1 + kt);
    }

    BFrag af[4], bf[2];
#pragma unroll
    for (int nt = 0; nt < 2; ++nt) {
      const unsigned short* pb = &sB[cur][(wn * 32 + nt * 16 + l16) * LDSS + 16 * lh];
      bf[nt].h[0] = *(const v8bf*)(pb);
      bf[nt].h[1] = *(const v8bf*)(pb + 8);
    }
#pragma unroll
    for (int mt = 0; mt < 4; ++mt) {
      const unsigned short* base = &sA[cur][(wm * 64 + mt * 16 + l16) * LDSS];
      af[mt].h[0] = *(const v8bf*)(base + 16 * lh);
      af[mt].h[1] = *(const v8bf*)(base + 16 * lh + 8);
#pragma unroll
      for (int nt = 0; nt < 2; ++nt)
        acc[mt][nt] = __builtin_amdgcn_wmma_f32_16x16x32_bf16(
            false, af[mt].v, false, bf[nt].v, (short)0, acc[mt][nt], false, false);
    }

    if (more) {
      const int nxt = cur ^ 1;
      *(uint4*)(&sA[nxt][ldsOff0]) = ra0;
      *(uint4*)(&sA[nxt][ldsOff1]) = ra1;
      *(uint4*)(&sB[nxt][ldsOff0]) = rb0;
      *(uint4*)(&sB[nxt][ldsOff1]) = rb1;
    }
  }

#pragma unroll
  for (int mt = 0; mt < 4; ++mt) {
#pragma unroll
    for (int v = 0; v < 8; ++v) {
      const int row = m0 + wm * 64 + mt * 16 + lh * 8 + v;
      const float wv = Wexp[(size_t)row * 8 + expert];
#pragma unroll
      for (int nt = 0; nt < 2; ++nt) {
        const int col = n0 + wn * 32 + nt * 16 + l16;
        const size_t o = (size_t)row * N + col;
        Out[o] += wv * acc[mt][nt][v];
      }
    }
  }
}

// ================================================================ host launcher
extern "C" void kernel_launch(void* const* d_in, const int* in_sizes, int n_in,
                              void* d_out, int out_size, void* d_ws, size_t ws_size,
                              hipStream_t stream) {
  (void)n_in; (void)ws_size;
  const float* x  = (const float*)d_in[0];   // [B,T,D] fp32
  const float* Wr = (const float*)d_in[1];   // [D,E]
  const float* Wg = (const float*)d_in[2];   // [E,D,F]
  const float* Wu = (const float*)d_in[3];   // [E,D,F]
  const float* Wd = (const float*)d_in[4];   // [E,F,D]
  float* out = (float*)d_out;

  const int D = 2048, F = 2048, E = 8;
  const int M = in_sizes[0] / D;             // B*T = 8192 tokens

  // workspace layout (bytes)
  char* ws = (char*)d_ws;
  unsigned short* Xb  = (unsigned short*)(ws);                                        // [M,D] bf16
  unsigned short* H   = (unsigned short*)(ws + (size_t)M * D * 2);                    // [M,F] bf16
  unsigned short* WgT = (unsigned short*)(ws + (size_t)M * D * 2 + (size_t)M * F * 2);// [F,D] bf16
  unsigned short* WuT = WgT + (size_t)F * D;                                          // [F,D] bf16
  unsigned short* WdT = WuT + (size_t)F * D;                                          // [D,F] bf16
  float* Wexp = (float*)(WdT + (size_t)D * F);                                        // [M,8] fp32

  // 1) zero output, convert X, route
  zero_f32_kernel<<<2048, 256, 0, stream>>>((float4*)out, (size_t)out_size / 4);
  cvt_bf16_kernel<<<2048, 256, 0, stream>>>(x, Xb, (size_t)M * D / 4);
  router_kernel<<<M / 8, 256, 0, stream>>>(x, Wr, Wexp, D);

  const dim3 tblk(32, 8);
  const dim3 ggrid(F / 128, M / 128);
  const dim3 dgrid(D / 128, M / 128);

  // 2) per expert: convert+transpose weights, fused gate/up, routed down-proj
  for (int e = 0; e < E; ++e) {
    transpose_cvt_kernel<<<dim3(F / 32, D / 32), tblk, 0, stream>>>(Wg + (size_t)e * D * F, WgT, D, F);
    transpose_cvt_kernel<<<dim3(F / 32, D / 32), tblk, 0, stream>>>(Wu + (size_t)e * D * F, WuT, D, F);
    transpose_cvt_kernel<<<dim3(D / 32, F / 32), tblk, 0, stream>>>(Wd + (size_t)e * F * D, WdT, F, D);
    moe_gateup_kernel<<<ggrid, 256, 0, stream>>>(Xb, WgT, WuT, H, M, F, D);
    moe_down_kernel<<<dgrid, 256, 0, stream>>>(H, WdT, Wexp, out, M, D, F, e);
  }
}
